// NewDnnPosTagger_53386443489827
// MI455X (gfx1250) — matrix operand order repaired and checked by reading, not compiled, not run on hardware
//
#include <hip/hip_runtime.h>
#include <hip/hip_bf16.h>

typedef __attribute__((ext_vector_type(16))) __bf16 v16bf;
typedef __attribute__((ext_vector_type(8)))  float  v8f;
typedef __attribute__((ext_vector_type(4)))  unsigned int u32x4;
typedef __attribute__((ext_vector_type(8)))  int i32x8;
typedef __attribute__((ext_vector_type(4)))  int i32x4;

#define Bsz   32
#define Lseq  100
#define Hh    100
#define G4    400      // 4*H
#define IN0P  128      // 120 padded to mult of 32
#define HP    128      // 100 padded (K extent for recurrent matmul)
#define HPL   136      // HP + 8 bf16 pad -> 272B row stride (bank-conflict free)
#define H2P   224      // 200 padded
#define MROWS (Bsz * Lseq)   // 3200
#define FCH   400

// ---------------- WMMA fragment loaders (wave32, ISA 7.12.2 layouts) -------
// A: 16x32 bf16 from [M][K] row-major. lane l: M=l&15;
//   lo lanes: e0..7->K=k+0..7,  e8..15->K=k+16..23 ; hi lanes shifted by 8.
__device__ __forceinline__ v16bf load_a_frag(const __bf16* base, int ldk, int kbase) {
  int lane = threadIdx.x & 31;
  const __bf16* p = base + (size_t)(lane & 15) * ldk + kbase + ((lane >> 4) << 3);
  union { v16bf v; uint4 q[2]; } u;
  u.q[0] = *(const uint4*)(p);
  u.q[1] = *(const uint4*)(p + 16);
  return u.v;
}
// B: 32x16 bf16 from [N][K] row-major. lane l: N=l&15; same K split as A (e0..7->K=k+0..7, e8..15->K=k+16..23; hi lanes +8).
__device__ __forceinline__ v16bf load_b_frag(const __bf16* base, int ldk, int kbase) {
  int lane = threadIdx.x & 31;
  const __bf16* p = base + (size_t)(lane & 15) * ldk + kbase + ((lane >> 4) << 3);
  union { v16bf v; uint4 q[2]; } u;
  u.q[0] = *(const uint4*)(p);
  u.q[1] = *(const uint4*)(p + 16);
  return u.v;
}

__device__ __forceinline__ float sigm(float x) { return 1.f / (1.f + __expf(-x)); }

// ---- Tensor Data Mover: 400x256B tile -> LDS with 16B row pad (272B stride)
__device__ __forceinline__ void tdm_load_whh(const __bf16* gsrc, void* lds_dst) {
  unsigned long long ga = (unsigned long long)(uintptr_t)gsrc;
  unsigned int lds = (unsigned int)(uintptr_t)lds_dst;   // low 32b = LDS offset
  u32x4 g0;
  g0[0] = 1u;                                            // count=1 user descriptor
  g0[1] = lds;                                           // lds_addr
  g0[2] = (unsigned int)ga;                              // global_addr[31:0]
  g0[3] = (unsigned int)((ga >> 32) & 0x1FFFFFFu) | (2u << 30); // [56:32] | type=2
  i32x8 g1;
  // data_size=8B(3), pad_enable, pad_interval=64DW(5), pad_amount=4DW(3)
  g1[0] = (3 << 16) | (1 << 20) | (5 << 22) | (3 << 25);
  g1[1] = (int)(32u << 16);        // tensor_dim0[15:0]=32 (8B units per row)
  g1[2] = (int)(400u << 16);       // tensor_dim0 hi=0 | tensor_dim1[15:0]=400
  g1[3] = (int)(32u << 16);        // tensor_dim1 hi=0 | tile_dim0=32
  g1[4] = 400;                     // tile_dim1=400 | tile_dim2=0
  g1[5] = 32;                      // tensor_dim0_stride=32
  g1[6] = 0; g1[7] = 0;
  i32x4 z4 = {0, 0, 0, 0};
#if defined(__clang_major__) && (__clang_major__ >= 23)
  i32x8 z8 = {0, 0, 0, 0, 0, 0, 0, 0};
  __builtin_amdgcn_tensor_load_to_lds(g0, g1, z4, z4, z8, 0);
#else
  __builtin_amdgcn_tensor_load_to_lds(g0, g1, z4, z4, 0);
#endif
}

// ---------------- small utility kernels ------------------------------------
__global__ void k_fill_zero16(unsigned short* p, int n) {
  int i = blockIdx.x * blockDim.x + threadIdx.x;
  if (i < n) p[i] = 0;
}

__global__ void k_build_x0(const float* __restrict__ word, const int* __restrict__ pidx,
                           const float* __restrict__ ptab, __bf16* __restrict__ X0) {
  int idx = blockIdx.x * blockDim.x + threadIdx.x;
  if (idx >= MROWS * IN0P) return;
  int row = idx >> 7;
  int k   = idx & 127;
  float v = 0.f;
  if (k < 100)      v = word[(size_t)row * 100 + k];
  else if (k < 120) v = ptab[(size_t)pidx[row] * 20 + (k - 100)];
  X0[idx] = (__bf16)v;
}

__global__ void k_convert_pad(const float* __restrict__ src, __bf16* __restrict__ dst,
                              int N, int K, int KP) {
  int idx = blockIdx.x * blockDim.x + threadIdx.x;
  if (idx >= N * KP) return;
  int n = idx / KP, k = idx - n * KP;
  dst[idx] = (k < K) ? (__bf16)src[(size_t)n * K + k] : (__bf16)0.f;
}

__global__ void k_transpose_pad(const float* __restrict__ src, __bf16* __restrict__ dst,
                                int K, int N, int KP) {
  int idx = blockIdx.x * blockDim.x + threadIdx.x;
  if (idx >= N * KP) return;
  int n = idx / KP, k = idx - n * KP;
  dst[idx] = (k < K) ? (__bf16)src[(size_t)k * N + n] : (__bf16)0.f;
}

// -------- bf16 WMMA GEMM, 1 M-tile x 5 N-tiles per wave (A-frag reuse) -----
// C[M][ldc] = A[M][KP] * B[N][KP]^T + bias0 + bias1    (N multiple of 80)
__global__ void __launch_bounds__(256)
k_gemm_bf16(const __bf16* __restrict__ A, const __bf16* __restrict__ Bm,
            float* __restrict__ C, int M, int N, int KP, int ldc,
            const float* __restrict__ bias0, const float* __restrict__ bias1) {
  constexpr int NT = 5;
  int wid  = threadIdx.x >> 5;
  int lane = threadIdx.x & 31;
  int ngroups = (N >> 4) / NT;
  int total   = (M >> 4) * ngroups;
  int w = blockIdx.x * (blockDim.x >> 5) + wid;
  if (w >= total) return;                        // wave-uniform exit
  int mt = w / ngroups, ng = w - mt * ngroups;
  const __bf16* Abase = A + (size_t)(mt * 16) * KP;
  v8f acc[NT] = {};
  for (int k = 0; k < KP; k += 32) {
    v16bf af = load_a_frag(Abase, KP, k);
#pragma unroll
    for (int j = 0; j < NT; ++j) {
      v16bf bf = load_b_frag(Bm + (size_t)((ng * NT + j) * 16) * KP, KP, k);
      acc[j] = __builtin_amdgcn_wmma_f32_16x16x32_bf16(false, af, false, bf,
                                                       (short)0, acc[j], false, false);
    }
  }
  int hi = lane >> 4;
#pragma unroll
  for (int j = 0; j < NT; ++j) {
    int n = (ng * NT + j) * 16 + (lane & 15);
    float bias = (bias0 ? bias0[n] : 0.f) + (bias1 ? bias1[n] : 0.f);
#pragma unroll
    for (int r = 0; r < 8; ++r)
      C[(size_t)(mt * 16 + r + hi * 8) * ldc + n] = acc[j][r] + bias;
  }
}

// ---------------- persistent recurrent scan: one block per direction -------
// xg: [2][B][L][400] fp32, Whh: [2][400][128] bf16 (global, TDM-staged to LDS)
__global__ void __launch_bounds__(512)
k_lstm_scan(const float* __restrict__ xg, const __bf16* __restrict__ Whh,
            __bf16* __restrict__ Y, int yld) {
  const int dir  = blockIdx.x;
  const int tid  = threadIdx.x;
  const int wid  = tid >> 5;
  const int lane = tid & 31;
  const float*  xgd = xg  + (size_t)dir * Bsz * Lseq * G4;
  const __bf16* whd = Whh + (size_t)dir * G4 * HP;
  const int colofs = dir * Hh;

  __shared__ __bf16 whh_s[G4][HPL];   // 108.8 KB, 272B row stride (no bank conflicts)
  __shared__ __bf16 h_s[Bsz][HPL];    // 8.7 KB
  __shared__ __bf16 g_s[Bsz][G4];     // 25.6 KB gate pre-activations

  if (wid == 0) {                     // one wave kicks the TDM and waits on it
    tdm_load_whh(whd, &whh_s[0][0]);
    __builtin_amdgcn_s_wait_tensorcnt(0);
  }
  float creg[7];
#pragma unroll
  for (int i = 0; i < 7; ++i) creg[i] = 0.f;
  for (int i = tid; i < Bsz * HPL; i += blockDim.x) (&h_s[0][0])[i] = (__bf16)0.f;
  __syncthreads();

  for (int t = 0; t < Lseq; ++t) {
    int tt = dir ? (Lseq - 1 - t) : t;
    // g = h @ Whh^T + xg[:, tt, :]  -- 2x25 tiles over 16 waves, all LDS operands
    for (int tile = wid; tile < 2 * 25; tile += 16) {
      int mt = tile / 25, nt = tile - mt * 25;
      v8f acc = {};
#pragma unroll
      for (int k = 0; k < HP; k += 32) {
        v16bf af = load_a_frag(&h_s[mt * 16][0], HPL, k);
        v16bf bf = load_b_frag(&whh_s[nt * 16][0], HPL, k);
        acc = __builtin_amdgcn_wmma_f32_16x16x32_bf16(false, af, false, bf,
                                                      (short)0, acc, false, false);
      }
      int hi = lane >> 4;
      int n  = nt * 16 + (lane & 15);
#pragma unroll
      for (int r = 0; r < 8; ++r) {
        int b = mt * 16 + r + hi * 8;
        g_s[b][n] = (__bf16)(acc[r] + xgd[((size_t)b * Lseq + tt) * G4 + n]);
      }
    }
    __syncthreads();
    // gate nonlinearities + state update (i,f,g,o order); c lives in registers
#pragma unroll
    for (int it = 0; it < 7; ++it) {
      int idx = tid + it * 512;
      if (idx < Bsz * Hh) {
        int b = idx / Hh, j = idx - b * Hh;
        float gi = (float)g_s[b][j],          gf = (float)g_s[b][j + Hh];
        float gg = (float)g_s[b][j + 2 * Hh], go = (float)g_s[b][j + 3 * Hh];
        float cn = sigm(gf) * creg[it] + sigm(gi) * tanhf(gg);
        float hv = sigm(go) * tanhf(cn);
        creg[it] = cn;
        __bf16 hb = (__bf16)hv;
        h_s[b][j] = hb;
        Y[((size_t)b * Lseq + tt) * yld + colofs + j] = hb;
        // prefetch next step's gate inputs (128B line coverage of the row)
        if (t + 1 < Lseq && (j & 31) == 0) {
          int tn = dir ? (Lseq - 2 - t) : (t + 1);
          const float* nx = &xgd[((size_t)b * Lseq + tn) * G4];
          __builtin_prefetch(nx + j,       0, 3);
          __builtin_prefetch(nx + j + 100, 0, 3);
          __builtin_prefetch(nx + j + 200, 0, 3);
          __builtin_prefetch(nx + j + 300, 0, 3);
        }
      }
    }
    __syncthreads();
  }
}

// ---------------- fused pairwise scorer ------------------------------------
__global__ void k_scores(const float* __restrict__ Ab, const float* __restrict__ Cb,
                         const float* __restrict__ w2, const float* __restrict__ b2,
                         const int* __restrict__ lens, float* __restrict__ out) {
  int bi = blockIdx.x;
  int b  = bi / (Lseq + 1);
  int i  = bi - b * (Lseq + 1);
  int tid = threadIdx.x;
  float* orow = out + ((size_t)b * (Lseq + 1) + i) * Lseq;
  if (i == Lseq) {
    for (int j = tid; j < Lseq; j += blockDim.x) orow[j] = 0.f;
    return;
  }
  __shared__ float arow[FCH];
  __shared__ float w2s[FCH];
  for (int k = tid; k < FCH; k += blockDim.x) {
    arow[k] = Ab[((size_t)b * Lseq + i) * FCH + k];
    w2s[k]  = w2[k];
  }
  __syncthreads();
  int len = lens[b];
  if (tid < Lseq) {
    int j = tid;
    float s = 0.f;
    if (i < len && j < len) {
      const float4* crow = (const float4*)(Cb + ((size_t)b * Lseq + j) * FCH);
      float acc = 0.f;
#pragma unroll 4
      for (int k4 = 0; k4 < FCH / 4; ++k4) {
        float4 c = crow[k4];
        acc += fmaxf(arow[4 * k4 + 0] + c.x, 0.f) * w2s[4 * k4 + 0]
             + fmaxf(arow[4 * k4 + 1] + c.y, 0.f) * w2s[4 * k4 + 1]
             + fmaxf(arow[4 * k4 + 2] + c.z, 0.f) * w2s[4 * k4 + 2]
             + fmaxf(arow[4 * k4 + 3] + c.w, 0.f) * w2s[4 * k4 + 3];
      }
      s = acc + b2[0];
    }
    orow[j] = s;
  }
}

// ---------------- host orchestration ---------------------------------------
extern "C" void kernel_launch(void* const* d_in, const int* in_sizes, int n_in,
                              void* d_out, int out_size, void* d_ws, size_t ws_size,
                              hipStream_t stream) {
  (void)in_sizes; (void)n_in; (void)out_size; (void)ws_size;
  const float* word  = (const float*)d_in[0];
  const int*   pidx  = (const int*)  d_in[1];
  const int*   lens  = (const int*)  d_in[2];
  const float* ptab  = (const float*)d_in[3];
  const float* Wih0  = (const float*)d_in[4];
  const float* Whh0  = (const float*)d_in[5];
  const float* bih0  = (const float*)d_in[6];
  const float* bhh0  = (const float*)d_in[7];
  const float* Wih   = (const float*)d_in[8];
  const float* Whh   = (const float*)d_in[9];
  const float* bih   = (const float*)d_in[10];
  const float* bhh   = (const float*)d_in[11];
  const float* W1a   = (const float*)d_in[12];
  const float* W1b   = (const float*)d_in[13];
  const float* b1    = (const float*)d_in[14];
  const float* W2    = (const float*)d_in[15];
  const float* b2    = (const float*)d_in[16];
  float* out = (float*)d_out;

  char* ws = (char*)d_ws;
  size_t off = 0;
  auto alloc = [&](size_t bytes) -> void* {
    void* p = ws + off;
    off = (off + bytes + 255) & ~(size_t)255;
    return p;
  };
  __bf16* X0   = (__bf16*)alloc((size_t)MROWS * IN0P * 2);
  __bf16* W0b  = (__bf16*)alloc((size_t)2 * G4 * IN0P * 2);
  __bf16* Whhb = (__bf16*)alloc((size_t)4 * 2 * G4 * HP * 2);
  __bf16* Wihb = (__bf16*)alloc((size_t)3 * 2 * G4 * H2P * 2);
  __bf16* W1aT = (__bf16*)alloc((size_t)FCH * H2P * 2);
  __bf16* W1bT = (__bf16*)alloc((size_t)FCH * H2P * 2);
  __bf16* Ya   = (__bf16*)alloc((size_t)MROWS * H2P * 2);
  __bf16* Yb   = (__bf16*)alloc((size_t)MROWS * H2P * 2);
  float*  xg   = (float*) alloc((size_t)2 * MROWS * G4 * 4);
  float*  Abuf = xg;                         // reuse xg region after scans
  float*  Cbuf = xg + (size_t)MROWS * G4;

  const int T = 256;
  auto cdiv = [](int a, int b) { return (a + b - 1) / b; };

  int nY = MROWS * H2P;
  k_fill_zero16<<<cdiv(nY, T), T, 0, stream>>>((unsigned short*)Ya, nY);
  k_fill_zero16<<<cdiv(nY, T), T, 0, stream>>>((unsigned short*)Yb, nY);

  int nX = MROWS * IN0P;
  k_build_x0<<<cdiv(nX, T), T, 0, stream>>>(word, pidx, ptab, X0);

  for (int d = 0; d < 2; ++d) {
    int n0 = G4 * IN0P;
    k_convert_pad<<<cdiv(n0, T), T, 0, stream>>>(Wih0 + (size_t)d * G4 * 120,
                                                 W0b + (size_t)d * G4 * IN0P, G4, 120, IN0P);
    int nh = G4 * HP;
    k_convert_pad<<<cdiv(nh, T), T, 0, stream>>>(Whh0 + (size_t)d * G4 * 100,
                                                 Whhb + (size_t)d * G4 * HP, G4, 100, HP);
  }
  for (int l = 0; l < 3; ++l)
    for (int d = 0; d < 2; ++d) {
      int ni = G4 * H2P;
      k_convert_pad<<<cdiv(ni, T), T, 0, stream>>>(Wih + (size_t)(l * 2 + d) * G4 * 200,
                                                   Wihb + (size_t)(l * 2 + d) * G4 * H2P,
                                                   G4, 200, H2P);
      int nh = G4 * HP;
      k_convert_pad<<<cdiv(nh, T), T, 0, stream>>>(Whh + (size_t)(l * 2 + d) * G4 * 100,
                                                   Whhb + (size_t)((l + 1) * 2 + d) * G4 * HP,
                                                   G4, 100, HP);
    }
  int nt1 = FCH * H2P;
  k_transpose_pad<<<cdiv(nt1, T), T, 0, stream>>>(W1a, W1aT, 200, FCH, H2P);
  k_transpose_pad<<<cdiv(nt1, T), T, 0, stream>>>(W1b, W1bT, 200, FCH, H2P);

  auto gemm = [&](const __bf16* A, const __bf16* Bm, float* C, int M, int N, int KP,
                  int ldc, const float* bb0, const float* bb1) {
    int waves = (M / 16) * ((N / 16) / 5);
    k_gemm_bf16<<<cdiv(waves, 8), 256, 0, stream>>>(A, Bm, C, M, N, KP, ldc, bb0, bb1);
  };

  // layer 0
  for (int d = 0; d < 2; ++d)
    gemm(X0, W0b + (size_t)d * G4 * IN0P, xg + (size_t)d * MROWS * G4,
         MROWS, G4, IN0P, G4, bih0 + d * G4, bhh0 + d * G4);
  k_lstm_scan<<<2, 512, 0, stream>>>(xg, Whhb, Ya, H2P);

  // layers 1..3
  __bf16* yin = Ya; __bf16* yout = Yb;
  for (int l = 0; l < 3; ++l) {
    for (int d = 0; d < 2; ++d)
      gemm(yin, Wihb + (size_t)(l * 2 + d) * G4 * H2P, xg + (size_t)d * MROWS * G4,
           MROWS, G4, H2P, G4, bih + (size_t)(l * 2 + d) * G4, bhh + (size_t)(l * 2 + d) * G4);
    k_lstm_scan<<<2, 512, 0, stream>>>(xg, Whhb + (size_t)(l + 1) * 2 * G4 * HP, yout, H2P);
    __bf16* tswap = yin; yin = yout; yout = tswap;
  }

  // pairwise projections: A' = Y@W1a + b1 ; C = Y@W1b   (reuse xg region)
  gemm(yin, W1aT, Abuf, MROWS, FCH, H2P, FCH, b1, nullptr);
  gemm(yin, W1bT, Cbuf, MROWS, FCH, H2P, FCH, nullptr, nullptr);

  k_scores<<<Bsz * (Lseq + 1), 128, 0, stream>>>(Abuf, Cbuf, W2, b2, lens, out);
}
